// binary_dense_35923106464285
// MI455X (gfx1250) — compile-verified, matrix operand order checked by reading, not run on hardware
//
#include <hip/hip_runtime.h>

typedef __attribute__((ext_vector_type(16))) _Float16 v16h;
typedef __attribute__((ext_vector_type(8)))  _Float16 v8h;
typedef __attribute__((ext_vector_type(8)))  float    v8f;
typedef __attribute__((ext_vector_type(4)))  float    v4f;
typedef __attribute__((ext_vector_type(4)))  int      v4i;

#define B_ROWS   65536
#define N_IN     1024
#define N_OUT    256
#define KCHUNK   32
#define NCHUNKS  (N_IN / KCHUNK)          // 32
#define M_TILE   128                      // rows per block (8 waves x 16)
#define LDS_ROW  40                       // halves per LDS row (80 B, padded from 64 B)
#define LDS_BUF  (N_OUT * LDS_ROW)        // halves per LDS buffer (20480 B)

#if __has_builtin(__builtin_amdgcn_global_load_async_to_lds_b128) && \
    __has_builtin(__builtin_amdgcn_s_wait_asynccnt)
#define HAVE_ASYNC 1
typedef __attribute__((address_space(1))) v4i as1_v4i;   // global int4
typedef __attribute__((address_space(3))) v4i as3_v4i;   // LDS int4 (32-bit ptr)
#else
#define HAVE_ASYNC 0
#endif

// ---------------------------------------------------------------------------
// Kernel 1: binarize weights to fp16 {-1,0,+1}, pre-packed chunk-major:
//   wPacked[(chunk*256 + n)*32 + (k%32)] = sign(w[k][n])
// so each K-chunk's 256x32 slab is a contiguous 16 KB block.
// ---------------------------------------------------------------------------
__global__ void __launch_bounds__(256) signpack_kernel(const float* __restrict__ w,
                                                       _Float16* __restrict__ wp) {
    int idx = blockIdx.x * 256 + threadIdx.x;   // 0 .. 1024*256-1
    int k = idx >> 8;                           // row in w   (0..1023)
    int n = idx & 255;                          // col in w   (0..255)
    float v = w[idx];
    float s = (v > 0.0f) ? 1.0f : ((v < 0.0f) ? -1.0f : 0.0f);
    int c   = k >> 5;
    int pos = k & 31;
    wp[(((size_t)c * 256 + n) << 5) + pos] = (_Float16)s;
}

// ---------------------------------------------------------------------------
// Stage one 256x32-half sign-weight slab (16 KB) into LDS buffer `buf`.
// Async path: global -> LDS direct (ASYNCcnt), no VGPR round-trip.
// ---------------------------------------------------------------------------
__device__ __forceinline__ void stage_chunk(const _Float16* __restrict__ Wp,
                                            _Float16* __restrict__ sbase,
                                            int c, int buf, int tid) {
    const _Float16* gsrc = Wp + (size_t)c * (N_OUT * KCHUNK);
    _Float16* dbase = sbase + buf * LDS_BUF;
    #pragma unroll
    for (int i = 0; i < 4; ++i) {
        int flat = i * 256 + tid;           // 0..1023, 16 B each
        int n    = flat >> 2;
        int part = flat & 3;
        const _Float16* src = gsrc + (size_t)flat * 8;
        _Float16*       dst = &dbase[n * LDS_ROW + part * 8];
#if HAVE_ASYNC
        __builtin_amdgcn_global_load_async_to_lds_b128(
            (as1_v4i*)(uintptr_t)src,
            (as3_v4i*)(unsigned)(uintptr_t)dst,
            0, 0);
#else
        *(v8h*)dst = *(const v8h*)src;      // global_load_b128 + ds_store_b128
#endif
    }
}

// ---------------------------------------------------------------------------
// Kernel 2: GEMM. Block = 256 thr (8 wave32). Block tile: 128 x 256 (full N).
// Wave computes 16 rows x 256 cols = 16 accumulator tiles (v8f each).
// Pipelines: LDS weight slabs double-buffered one chunk ahead (ASYNCcnt),
// A-fragment global loads one chunk ahead, B fragments one tile ahead.
// ---------------------------------------------------------------------------
__global__ void __launch_bounds__(256) binary_dense_gemm_kernel(
        const float*    __restrict__ X,
        const _Float16* __restrict__ Wp,
        float*          __restrict__ Y) {

    __shared__ __align__(16) _Float16 smem[2 * LDS_BUF];  // 40960 B

    const int tid    = threadIdx.x;
    const int lane   = tid & 31;
    const int wave   = tid >> 5;
    const int laneHi = lane >> 4;           // 0: lanes 0-15, 1: lanes 16-31
    const int laneLo = lane & 15;

    const int mBase = blockIdx.x * M_TILE + wave * 16;
    const float* xrow = X + (size_t)(mBase + laneLo) * N_IN;
    const int kLane = laneHi * 8;           // lane-half K offset within chunk

    v8f acc[16] = {};                       // 16 N-tiles of 16x16 f32

    // ---- prologue: stage chunk 0 weights; issue chunk 0 A loads ----
    stage_chunk(Wp, smem, 0, 0, tid);
    v4f p0 = *(const v4f*)(xrow + kLane);
    v4f p1 = *(const v4f*)(xrow + kLane + 4);
    v4f p2 = *(const v4f*)(xrow + kLane + 16);
    v4f p3 = *(const v4f*)(xrow + kLane + 20);

    for (int c = 0; c < NCHUNKS; ++c) {
        const int cur = c & 1;
#if HAVE_ASYNC
        __builtin_amdgcn_s_wait_asynccnt(0);   // my slab writes for `cur` done
#endif
        __syncthreads();                       // all waves' slab writes visible

        // stage NEXT chunk's weights into the other buffer (overlaps compute)
        if (c + 1 < NCHUNKS)
            stage_chunk(Wp, smem, c + 1, cur ^ 1, tid);

        // ---- convert current A chunk: fp32 -> (hi,lo) f16 split ----
        // A lane layout (ISA 7.12.2, 16x32 f16):
        //   lanes 0-15: K {0..7,16..23}; lanes 16-31: K {8..15,24..31}
        float xs[16];
        #pragma unroll
        for (int j = 0; j < 4; ++j) {
            xs[j]      = p0[j];
            xs[j + 4]  = p1[j];
            xs[j + 8]  = p2[j];
            xs[j + 12] = p3[j];
        }
        v16h ahi, alo;
        #pragma unroll
        for (int j = 0; j < 16; ++j) {
            _Float16 h = (_Float16)xs[j];
            ahi[j] = h;
            alo[j] = (_Float16)(xs[j] - (float)h);
        }

        // ---- issue NEXT chunk's A loads (consumed next iteration) ----
        if (c + 1 < NCHUNKS) {
            const int k0n = (c + 1) * KCHUNK + kLane;
            __builtin_prefetch(xrow + k0n + KCHUNK, 0, 0);  // global_prefetch_b8
            p0 = *(const v4f*)(xrow + k0n);
            p1 = *(const v4f*)(xrow + k0n + 4);
            p2 = *(const v4f*)(xrow + k0n + 16);
            p3 = *(const v4f*)(xrow + k0n + 20);
        }

        // ---- 16 N-tiles; B fragments double-buffered in registers ----
        const _Float16* sb = smem + cur * LDS_BUF;
        auto loadB = [&](int t) -> v16h {
            const _Float16* bp = &sb[(t * 16 + laneLo) * LDS_ROW + laneHi * 8];
            v8h b0 = *(const v8h*)(bp);          // ds_load_b128
            v8h b1 = *(const v8h*)(bp + 16);     // ds_load_b128
            return __builtin_shufflevector(b0, b1,
                     0, 1, 2, 3, 4, 5, 6, 7, 8, 9, 10, 11, 12, 13, 14, 15);
        };

        v16h bcur = loadB(0);
        #pragma unroll
        for (int t = 0; t < 16; ++t) {
            v16h bnext = (t < 15) ? loadB(t + 1) : bcur;  // issued before WMMAs
            acc[t] = __builtin_amdgcn_wmma_f32_16x16x32_f16(
                        false, ahi, false, bcur, (short)0, acc[t], false, false);
            acc[t] = __builtin_amdgcn_wmma_f32_16x16x32_f16(
                        false, alo, false, bcur, (short)0, acc[t], false, false);
            bcur = bnext;
        }
    }

    // ---- store: C/D layout (ISA 7.12.2): VGPR r -> (M=r | M=8+r), N=lane ----
    float* yb = Y + (size_t)mBase * N_OUT;
    #pragma unroll
    for (int t = 0; t < 16; ++t) {
        #pragma unroll
        for (int r = 0; r < 8; ++r) {
            int row = r + laneHi * 8;
            yb[(size_t)row * N_OUT + t * 16 + laneLo] = acc[t][r];
        }
    }
}

// ---------------------------------------------------------------------------
extern "C" void kernel_launch(void* const* d_in, const int* in_sizes, int n_in,
                              void* d_out, int out_size, void* d_ws, size_t ws_size,
                              hipStream_t stream) {
    const float* x = (const float*)d_in[0];      // [65536, 1024] fp32
    const float* w = (const float*)d_in[1];      // [1024, 256]  fp32
    float* y = (float*)d_out;                    // [65536, 256] fp32
    _Float16* wPacked = (_Float16*)d_ws;         // 1024*256 halves = 512 KB

    signpack_kernel<<<(N_IN * N_OUT) / 256, 256, 0, stream>>>(w, wPacked);
    binary_dense_gemm_kernel<<<B_ROWS / M_TILE, 256, 0, stream>>>(x, wPacked, y);
}
